// MultiHeadSequenceAttention_19499151524217
// MI455X (gfx1250) — matrix operand order repaired and checked by reading, not compiled, not run on hardware
//
#include <hip/hip_runtime.h>
#include <cstdint>
#include <cstddef>

// ---------------------------------------------------------------------------
// Problem constants (B,S,D,H,HD from reference)
// ---------------------------------------------------------------------------
static constexpr int BATCH = 2;
static constexpr int SEQ   = 2048;
static constexpr int DIM   = 1024;
static constexpr int NH    = 16;
static constexpr int HD    = 64;
static constexpr int MROWS = BATCH * SEQ;   // 4096

typedef __attribute__((ext_vector_type(16))) __bf16 v16bf;
typedef __attribute__((ext_vector_type(8)))  __bf16 v8bf;
typedef __attribute__((ext_vector_type(8)))  float  v8f;
typedef __attribute__((ext_vector_type(4)))  unsigned int v4u;
typedef __attribute__((ext_vector_type(8)))  int v8i;
typedef __attribute__((ext_vector_type(4)))  int v4i;

#define DEV __device__ __forceinline__

#if __has_builtin(__builtin_amdgcn_tensor_load_to_lds)
#define HAVE_TDM 1
#else
#define HAVE_TDM 0
#endif

DEV __bf16 f2bf(float f) {
  uint32_t u = __builtin_bit_cast(uint32_t, f);
  u += 0x7FFFu + ((u >> 16) & 1u);           // round to nearest even
  return __builtin_bit_cast(__bf16, (uint16_t)(u >> 16));
}

DEV v8f wmma_bf16(v16bf a, v16bf b, v8f c) {
  // v_wmma_f32_16x16x32_bf16  D = A(16x32) * B(32x16) + C(16x16 f32)
  return __builtin_amdgcn_wmma_f32_16x16x32_bf16(
      /*neg_a=*/false, a, /*neg_b=*/false, b,
      /*c_mod=*/(short)0, c, /*reuse_a=*/false, /*reuse_b=*/false);
}

// A-operand fragment (16x32 bf16). Lane holds row M = lane&15.
// ISA A layout: elems 0..7 -> K = half*8..+7 ; elems 8..15 -> K = 16+half*8..+7
DEV v16bf load_frag_a(const __bf16* row, int half) {
  v8bf c0 = *(const v8bf*)(row + half * 8);
  v8bf c1 = *(const v8bf*)(row + 16 + half * 8);
  v16bf f;
#pragma unroll
  for (int i = 0; i < 8; ++i) { f[i] = c0[i]; f[i + 8] = c1[i]; }
  return f;
}

// B-operand fragment (32x16 bf16), supplied as B^T rows (row = column N).
// ISA B layout: lane holds column N = lane&15, elems 0..7 -> K = half*8..+7 ; elems 8..15 -> K = 16+half*8..+7
DEV v16bf load_frag_b(const __bf16* row, int half) {
  v8bf c0 = *(const v8bf*)(row + half * 8);
  v8bf c1 = *(const v8bf*)(row + 16 + half * 8);
  v16bf f;
#pragma unroll
  for (int i = 0; i < 8; ++i) { f[i] = c0[i]; f[i + 8] = c1[i]; }
  return f;
}

// Row reductions across the 16 lanes sharing a C-matrix row, via ds_swizzle
// group-of-32 xor mode (SWAPX1/2/4/8): offset = {xor[14:10], or[9:5], and[4:0]}
DEV float lane_xor_f(float v, int pattern) {
#if __has_builtin(__builtin_amdgcn_ds_swizzle)
  switch (pattern) {   // ds_swizzle needs an immediate
    case 1:  return __builtin_bit_cast(float, __builtin_amdgcn_ds_swizzle(__builtin_bit_cast(int, v), 0x041F));
    case 2:  return __builtin_bit_cast(float, __builtin_amdgcn_ds_swizzle(__builtin_bit_cast(int, v), 0x081F));
    case 4:  return __builtin_bit_cast(float, __builtin_amdgcn_ds_swizzle(__builtin_bit_cast(int, v), 0x101F));
    default: return __builtin_bit_cast(float, __builtin_amdgcn_ds_swizzle(__builtin_bit_cast(int, v), 0x201F));
  }
#else
  return __shfl_xor(v, pattern, 32);
#endif
}
DEV float rmax16(float v) {
  v = fmaxf(v, lane_xor_f(v, 1));
  v = fmaxf(v, lane_xor_f(v, 2));
  v = fmaxf(v, lane_xor_f(v, 4));
  v = fmaxf(v, lane_xor_f(v, 8));
  return v;
}
DEV float rsum16(float v) {
  v += lane_xor_f(v, 1);
  v += lane_xor_f(v, 2);
  v += lane_xor_f(v, 4);
  v += lane_xor_f(v, 8);
  return v;
}

// ---------------------------------------------------------------------------
// Tensor Data Mover: 2D tile load global -> LDS (D# per CDNA5 ISA).
// data_size = 2 bytes (bf16). Padding adds pad_amount DWORDs to the LDS
// address every pad_interval DWORDs (row padding for bank-conflict-free
// fragment reads).  Issued by one wave; completion via s_wait_tensorcnt.
// This toolchain declares the 6-arg builtin:
//   (uint32x4 g0, int32x8 g1, int32x4 g2, int32x4 g3, int32x8 extra, i32 cpol)
// ---------------------------------------------------------------------------
DEV uint32_t lds_offset(const void* p) { return (uint32_t)(uintptr_t)p; }

DEV void tdm_load_2d_bf16(const void* gptr, uint32_t lds_addr,
                          uint32_t tensor_d0, uint32_t tensor_d1,
                          uint32_t stride_d0,
                          uint32_t tile_d0, uint32_t tile_d1,
                          uint32_t pad_interval_code, uint32_t pad_amount_code) {
#if HAVE_TDM
  uint64_t ga = (uint64_t)(uintptr_t)gptr;
  v4u g0;
  g0[0] = 1u;                                    // count=1, user descriptor
  g0[1] = lds_addr;                              // [63:32] lds_addr
  g0[2] = (uint32_t)ga;                          // [95:64] global_addr lo
  g0[3] = (uint32_t)(ga >> 32) | (2u << 30);     // [120:96] hi | type=2
  v8i g1;
  g1[0] = (int)((1u << 16)                       // data_size = 2B
              | (1u << 20)                       // pad_enable
              | (pad_interval_code << 22)
              | (pad_amount_code << 25));
  g1[1] = (int)((tensor_d0 & 0xFFFFu) << 16);    // tensor_dim0[15:0]
  g1[2] = (int)((tensor_d0 >> 16) | ((tensor_d1 & 0xFFFFu) << 16));
  g1[3] = (int)((tensor_d1 >> 16) | (tile_d0 << 16));
  g1[4] = (int)(tile_d1 & 0xFFFFu);              // tile_dim1, tile_dim2=0
  g1[5] = (int)stride_d0;                        // tensor_dim0_stride lo
  g1[6] = 0;                                     // stride hi, dim1_stride lo
  g1[7] = 0;
  v4i z4 = {0, 0, 0, 0};
  v8i z8 = {0, 0, 0, 0, 0, 0, 0, 0};
  __builtin_amdgcn_tensor_load_to_lds(g0, g1, z4, z4, z8, 0);
#endif
}

// ---------------------------------------------------------------------------
// fp32 -> bf16 bulk convert (X)
// ---------------------------------------------------------------------------
__global__ __launch_bounds__(256)
void f32_to_bf16_kernel(const float* __restrict__ in, __bf16* __restrict__ out) {
  size_t i = ((size_t)blockIdx.x * 256 + threadIdx.x) * 8;
  float4 a = *(const float4*)(in + i);
  float4 b = *(const float4*)(in + i + 4);
  v8bf o;
  o[0] = f2bf(a.x); o[1] = f2bf(a.y); o[2] = f2bf(a.z); o[3] = f2bf(a.w);
  o[4] = f2bf(b.x); o[5] = f2bf(b.y); o[6] = f2bf(b.z); o[7] = f2bf(b.w);
  *(v8bf*)(out + i) = o;
}

// ---------------------------------------------------------------------------
// W [K x N] fp32  ->  W^T [N x K] bf16   (64x64 LDS tile transpose)
// ---------------------------------------------------------------------------
__global__ __launch_bounds__(256)
void transpose_w_kernel(const float* __restrict__ W, __bf16* __restrict__ WT,
                        int K, int N) {
  __shared__ float tile[64][65];
  const int k0 = blockIdx.x * 64;
  const int n0 = blockIdx.y * 64;
  const int tid = threadIdx.x;
#pragma unroll
  for (int it = 0; it < 4; ++it) {            // 64 rows x 16 float4 chunks
    int cid = tid + it * 256;
    int r = cid >> 4, c = (cid & 15) * 4;
    float4 v = *(const float4*)(W + (size_t)(k0 + r) * N + n0 + c);
    tile[r][c] = v.x; tile[r][c + 1] = v.y; tile[r][c + 2] = v.z; tile[r][c + 3] = v.w;
  }
  __syncthreads();
#pragma unroll
  for (int it = 0; it < 2; ++it) {            // 64 n-rows x 8 v8bf chunks
    int cid = tid + it * 256;
    int n = cid >> 3, c = (cid & 7) * 8;
    v8bf o;
#pragma unroll
    for (int i = 0; i < 8; ++i) o[i] = f2bf(tile[c + i][n]);
    *(v8bf*)(WT + (size_t)(n0 + n) * K + k0 + c) = o;
  }
}

// ---------------------------------------------------------------------------
// bf16 GEMM:  C[M,N] = A[M,K] * B[K,N] + bias[N],  B given transposed (BT[N,K])
// Block tile 128x128, 8 waves (2x4), wave tile 64x32 = 4x2 WMMA accumulators,
// K-step 32.  A/B tiles staged in LDS by the Tensor Data Mover (wave 0 issues
// two tensor_load_to_lds, waits TENSORcnt, barrier releases the block).
// LDS row stride = 40 bf16 (80B, 16B multiple): tile row 64B + 16B pad
// expressed as D# pad_interval=16DW (code 3), pad_amount=4DW (code 3).
// ---------------------------------------------------------------------------
template <typename OutT>
__global__ __launch_bounds__(256)
void gemm_bf16_kernel(const __bf16* __restrict__ A, const __bf16* __restrict__ BT,
                      const float* __restrict__ bias, OutT* __restrict__ C,
                      int M, int N, int K) {
  constexpr int BM = 128, BN = 128, KB = 32, LS = KB + 8;
  __shared__ __align__(16) __bf16 sA[BM * LS];
  __shared__ __align__(16) __bf16 sB[BN * LS];

  const int tid  = threadIdx.x;
  const int lane = tid & 31, wave = tid >> 5;
  const int l15  = lane & 15, half = lane >> 4;
  const int wm   = (wave & 1) * 64;
  const int wn   = (wave >> 1) * 32;
  const int m0   = blockIdx.x * BM;
  const int n0   = blockIdx.y * BN;

  const v8f vzero = {0.f, 0.f, 0.f, 0.f, 0.f, 0.f, 0.f, 0.f};
  v8f acc[4][2];
#pragma unroll
  for (int mt = 0; mt < 4; ++mt)
#pragma unroll
    for (int nt = 0; nt < 2; ++nt) acc[mt][nt] = vzero;

  for (int k0 = 0; k0 < K; k0 += KB) {
    __syncthreads();                          // prior fragment reads complete
#if HAVE_TDM
    if (wave == 0) {
      tdm_load_2d_bf16(A + (size_t)m0 * K + k0, lds_offset(sA),
                       (uint32_t)K, (uint32_t)M, (uint32_t)K,
                       KB, BM, /*pad_int 16DW*/3, /*pad_amt 4DW*/3);
      tdm_load_2d_bf16(BT + (size_t)n0 * K + k0, lds_offset(sB),
                       (uint32_t)K, (uint32_t)N, (uint32_t)K,
                       KB, BN, 3, 3);
      __builtin_amdgcn_s_wait_tensorcnt(0);
    }
#else
#pragma unroll
    for (int it = 0; it < 2; ++it) {          // 128 rows x 4 chunks, 2/thread
      int cid = tid + it * 256;
      int row = cid >> 2, cc = (cid & 3) * 8;
      *(v8bf*)(sA + row * LS + cc) =
          *(const v8bf*)(A + (size_t)(m0 + row) * K + k0 + cc);
      *(v8bf*)(sB + row * LS + cc) =
          *(const v8bf*)(BT + (size_t)(n0 + row) * K + k0 + cc);
    }
#endif
    __syncthreads();                          // tile visible to all waves

    v16bf af[4], bfr[2];
#pragma unroll
    for (int mt = 0; mt < 4; ++mt)
      af[mt] = load_frag_a(sA + (size_t)(wm + mt * 16 + l15) * LS, half);
#pragma unroll
    for (int nt = 0; nt < 2; ++nt)
      bfr[nt] = load_frag_b(sB + (size_t)(wn + nt * 16 + l15) * LS, half);
#pragma unroll
    for (int mt = 0; mt < 4; ++mt)
#pragma unroll
      for (int nt = 0; nt < 2; ++nt)
        acc[mt][nt] = wmma_bf16(af[mt], bfr[nt], acc[mt][nt]);
  }

  // Epilogue: C layout -> element (M = r + 8*half, N = l15) per 16x16 tile
#pragma unroll
  for (int nt = 0; nt < 2; ++nt) {
    int gn = n0 + wn + nt * 16 + l15;
    float bv = bias[gn];
#pragma unroll
    for (int mt = 0; mt < 4; ++mt) {
#pragma unroll
      for (int r = 0; r < 8; ++r) {
        int gm = m0 + wm + mt * 16 + r + 8 * half;
        float v = acc[mt][nt][r] + bv;
        if constexpr (__is_same(OutT, float))
          C[(size_t)gm * N + gn] = v;
        else
          C[(size_t)gm * N + gn] = f2bf(v);
      }
    }
  }
}

// ---------------------------------------------------------------------------
// V [b*S, H*HD] bf16  ->  Vt [b, h, HD, S] bf16   (64x64 LDS tile per block)
// ---------------------------------------------------------------------------
__global__ __launch_bounds__(256)
void transpose_v_kernel(const __bf16* __restrict__ Vb, __bf16* __restrict__ Vt) {
  constexpr int TS = 72;                       // 144B rows: 16B multiple
  __shared__ __align__(16) __bf16 tile[64 * TS];
  const int bh = blockIdx.x >> 5;              // 0..31  (b*NH + h)
  const int s0 = (blockIdx.x & 31) * 64;
  const int b  = bh >> 4, h = bh & 15;
  const int tid = threadIdx.x;
#pragma unroll
  for (int it = 0; it < 2; ++it) {             // 64 s-rows x 8 v8bf chunks
    int cid = tid + it * 256;
    int sr = cid >> 3, c = (cid & 7) * 8;
    *(v8bf*)(tile + sr * TS + c) =
        *(const v8bf*)(Vb + (size_t)(b * SEQ + s0 + sr) * DIM + h * HD + c);
  }
  __syncthreads();
#pragma unroll
  for (int it = 0; it < 2; ++it) {             // 64 d-rows x 8 v8bf chunks
    int cid = tid + it * 256;
    int d = cid >> 3, c = (cid & 7) * 8;
    v8bf o;
#pragma unroll
    for (int i = 0; i < 8; ++i) o[i] = tile[(c + i) * TS + d];
    *(v8bf*)(Vt + (size_t)(bh * HD + d) * SEQ + s0 + c) = o;
  }
}

// ---------------------------------------------------------------------------
// Flash attention (non-causal). Grid = B*NH*(SEQ/128) blocks, 8 waves/block,
// each wave owns a 16-row query tile; all waves share a (b,h) so the K-tile
// (32x64) and V-tile (64x32, from Vt) are TDM-staged into LDS once per block
// per j-step instead of 8x redundant global loads. P is restaged through
// per-wave LDS from C-layout into A-fragment layout for the P@V WMMAs.
// ---------------------------------------------------------------------------
__global__ __launch_bounds__(256)
void flash_attn_kernel(const __bf16* __restrict__ Qb, const __bf16* __restrict__ Kb,
                       const __bf16* __restrict__ Vt, __bf16* __restrict__ AO) {
  constexpr int PS = 40;                       // P stage stride (80B rows)
  constexpr int KS = 72;                       // K tile stride: 64+8 (144B)
  constexpr int VS = 40;                       // V tile stride: 32+8 (80B)
  __shared__ __align__(16) __bf16 Kt[32 * KS];
  __shared__ __align__(16) __bf16 Vtl[64 * VS];
  __shared__ __align__(16) __bf16 Pst[8 * 16 * PS];

  const int tid = threadIdx.x, lane = tid & 31, wave = tid >> 5;
  const int l15 = lane & 15, half = lane >> 4;
  const int qblk = blockIdx.x & 15;            // SEQ/128 = 16
  const int bh   = blockIdx.x >> 4;            // 0..31
  const int b = bh >> 4, h = bh & 15;
  const int q0 = qblk * 128 + wave * 16;

  const __bf16* qrow = Qb + (size_t)(b * SEQ + q0 + l15) * DIM + h * HD;
  v16bf qa0 = load_frag_a(qrow, half);         // d = 0..31
  v16bf qa1 = load_frag_a(qrow + 32, half);    // d = 32..63

  __bf16* pw = Pst + wave * 16 * PS;
  const v8f vzero = {0.f, 0.f, 0.f, 0.f, 0.f, 0.f, 0.f, 0.f};
  v8f o0 = vzero, o1 = vzero, o2 = vzero, o3 = vzero;
  float mrow[8], lrow[8];
#pragma unroll
  for (int r = 0; r < 8; ++r) { mrow[r] = -1e30f; lrow[r] = 0.f; }

  for (int j = 0; j < SEQ; j += 32) {
    __syncthreads();                           // prior K/V tile reads complete
#if HAVE_TDM
    if (wave == 0) {
      // K tile: 32 key rows x 64 d (row 128B + 16B pad -> int=32DW(4), amt=4DW(3))
      tdm_load_2d_bf16(Kb + (size_t)(b * SEQ + j) * DIM + h * HD, lds_offset(Kt),
                       DIM, MROWS, DIM, 64, 32, 4, 3);
      // V tile: 64 d rows x 32 keys (row 64B + 16B pad -> int=16DW(3), amt=4DW(3))
      tdm_load_2d_bf16(Vt + (size_t)(bh * HD) * SEQ + j, lds_offset(Vtl),
                       SEQ, BATCH * NH * HD, SEQ, 32, 64, 3, 3);
      __builtin_amdgcn_s_wait_tensorcnt(0);
    }
#else
    {
      int r = tid >> 3, c = (tid & 7) * 8;     // K: 32 rows x 8 chunks
      *(v8bf*)(Kt + r * KS + c) =
          *(const v8bf*)(Kb + (size_t)(b * SEQ + j + r) * DIM + h * HD + c);
      int r2 = tid >> 2, c2 = (tid & 3) * 8;   // V: 64 rows x 4 chunks
      *(v8bf*)(Vtl + r2 * VS + c2) =
          *(const v8bf*)(Vt + (size_t)(bh * HD + r2) * SEQ + j + c2);
    }
#endif
    __syncthreads();                           // tiles visible

    const __bf16* klo = Kt + (size_t)l15 * KS;
    const __bf16* khi = Kt + (size_t)(16 + l15) * KS;
    v8f sl = vzero, sh = vzero;
    sl = wmma_bf16(qa0, load_frag_b(klo, half), sl);
    sl = wmma_bf16(qa1, load_frag_b(klo + 32, half), sl);
    sh = wmma_bf16(qa0, load_frag_b(khi, half), sh);
    sh = wmma_bf16(qa1, load_frag_b(khi + 32, half), sh);

#pragma unroll
    for (int r = 0; r < 8; ++r) {
      float a = sl[r] * 0.125f;                // 1/sqrt(HD)
      float c = sh[r] * 0.125f;
      float mnew = fmaxf(mrow[r], rmax16(fmaxf(a, c)));
      float corr = __expf(mrow[r] - mnew);
      float pa = __expf(a - mnew);
      float pb = __expf(c - mnew);
      lrow[r] = lrow[r] * corr + rsum16(pa + pb);
      mrow[r] = mnew;
      o0[r] *= corr; o1[r] *= corr; o2[r] *= corr; o3[r] *= corr;
      __bf16* prow = pw + (size_t)(r + 8 * half) * PS;   // C row -> P row
      prow[l15]      = f2bf(pa);               // keys j    .. j+15
      prow[16 + l15] = f2bf(pb);               // keys j+16 .. j+31
    }
    // P staging is per-wave private: only in-wave LDS ordering needed
    asm volatile("s_wait_dscnt 0" ::: "memory");

    v16bf pf = load_frag_a(pw + (size_t)l15 * PS, half);
    const __bf16* vb0 = Vtl + (size_t)l15 * VS;
    o0 = wmma_bf16(pf, load_frag_b(vb0, half), o0);
    o1 = wmma_bf16(pf, load_frag_b(vb0 + (size_t)16 * VS, half), o1);
    o2 = wmma_bf16(pf, load_frag_b(vb0 + (size_t)32 * VS, half), o2);
    o3 = wmma_bf16(pf, load_frag_b(vb0 + (size_t)48 * VS, half), o3);
  }

  __bf16* orow = AO + (size_t)(b * SEQ + q0) * DIM + h * HD;
#pragma unroll
  for (int r = 0; r < 8; ++r) {
    float inv = 1.0f / lrow[r];
    size_t ro = (size_t)(r + 8 * half) * DIM;
    orow[ro + l15]      = f2bf(o0[r] * inv);
    orow[ro + 16 + l15] = f2bf(o1[r] * inv);
    orow[ro + 32 + l15] = f2bf(o2[r] * inv);
    orow[ro + 48 + l15] = f2bf(o3[r] * inv);
  }
}

// ---------------------------------------------------------------------------
// Host launcher
// ---------------------------------------------------------------------------
extern "C" void kernel_launch(void* const* d_in, const int* in_sizes, int n_in,
                              void* d_out, int out_size, void* d_ws, size_t ws_size,
                              hipStream_t stream) {
  const float* X  = (const float*)d_in[0];
  const float* Wq = (const float*)d_in[1];
  const float* bq = (const float*)d_in[2];
  const float* Wk = (const float*)d_in[3];
  const float* bk = (const float*)d_in[4];
  const float* Wv = (const float*)d_in[5];
  const float* bv = (const float*)d_in[6];
  const float* Wo = (const float*)d_in[7];
  const float* bo = (const float*)d_in[8];
  float* out = (float*)d_out;

  char* ws = (char*)d_ws;
  size_t off = 0;
  auto carve = [&](size_t bytes) -> void* {
    void* p = ws + off;
    off += (bytes + 255) & ~(size_t)255;
    return p;
  };
  __bf16* Xb  = (__bf16*)carve((size_t)MROWS * DIM * 2);
  __bf16* WqT = (__bf16*)carve((size_t)DIM * DIM * 2);
  __bf16* WkT = (__bf16*)carve((size_t)DIM * DIM * 2);
  __bf16* WvT = (__bf16*)carve((size_t)DIM * DIM * 2);
  __bf16* WoT = (__bf16*)carve((size_t)DIM * DIM * 2);
  __bf16* Qb  = (__bf16*)carve((size_t)MROWS * DIM * 2);
  __bf16* Kb  = (__bf16*)carve((size_t)MROWS * DIM * 2);
  __bf16* Vb  = (__bf16*)carve((size_t)MROWS * DIM * 2);
  __bf16* Vt  = (__bf16*)carve((size_t)MROWS * DIM * 2);
  __bf16* AO  = (__bf16*)carve((size_t)MROWS * DIM * 2);

  // 1) convert X to bf16
  f32_to_bf16_kernel<<<(MROWS * DIM) / (256 * 8), 256, 0, stream>>>(X, Xb);

  // 2) transpose+convert the four weight matrices
  dim3 tg(DIM / 64, DIM / 64);
  transpose_w_kernel<<<tg, 256, 0, stream>>>(Wq, WqT, DIM, DIM);
  transpose_w_kernel<<<tg, 256, 0, stream>>>(Wk, WkT, DIM, DIM);
  transpose_w_kernel<<<tg, 256, 0, stream>>>(Wv, WvT, DIM, DIM);
  transpose_w_kernel<<<tg, 256, 0, stream>>>(Wo, WoT, DIM, DIM);

  // 3) Q/K/V projections (bf16 out, fp32 accumulate + bias)
  dim3 gg(MROWS / 128, DIM / 128);
  gemm_bf16_kernel<__bf16><<<gg, 256, 0, stream>>>(Xb, WqT, bq, Qb, MROWS, DIM, DIM);
  gemm_bf16_kernel<__bf16><<<gg, 256, 0, stream>>>(Xb, WkT, bk, Kb, MROWS, DIM, DIM);
  gemm_bf16_kernel<__bf16><<<gg, 256, 0, stream>>>(Xb, WvT, bv, Vb, MROWS, DIM, DIM);

  // 4) V -> [b,h,HD,S] so P@V B-fragments are contiguous
  transpose_v_kernel<<<BATCH * NH * (SEQ / 64), 256, 0, stream>>>(Vb, Vt);

  // 5) attention
  flash_attn_kernel<<<BATCH * NH * (SEQ / 128), 256, 0, stream>>>(Qb, Kb, Vt, AO);

  // 6) output projection, fp32 out + bias
  gemm_bf16_kernel<float><<<gg, 256, 0, stream>>>(AO, WoT, bo, out, MROWS, DIM, DIM);
}
